// MixtureOfExperts_20246475833418
// MI455X (gfx1250) — compile-verified, matrix operand order checked by reading, not run on hardware
//
#include <hip/hip_runtime.h>

// ---------------------------------------------------------------------------
// Types for CDNA5 WMMA (wave32): v_wmma_f32_16x16x32_bf16
// ---------------------------------------------------------------------------
typedef __attribute__((ext_vector_type(16))) __bf16 v16bf;
typedef __attribute__((ext_vector_type(8)))  __bf16 v8bf;
typedef __attribute__((ext_vector_type(8)))  float  v8f;

#define MOE_B   8192
#define MOE_M   16
#define MOE_N   8

// ---------------------------------------------------------------------------
// bf16 helpers (bit-level, round-to-nearest-even)
// ---------------------------------------------------------------------------
__device__ __forceinline__ unsigned short f32_to_bf16(float f) {
    unsigned int u = __builtin_bit_cast(unsigned int, f);
    u += 0x7FFFu + ((u >> 16) & 1u);
    return (unsigned short)(u >> 16);
}
__device__ __forceinline__ float bf16_to_f32(unsigned short h) {
    unsigned int u = ((unsigned int)h) << 16;
    return __builtin_bit_cast(float, u);
}

// Load one 16-bit A/B WMMA fragment for this lane.
// Per CDNA5 ISA: lanes 0-15 hold K chunks {k0..k0+7, k0+16..k0+23},
// lanes 16-31 hold {k0+8..k0+15, k0+24..k0+31}.  `p` already points at
// rowBase + k0 + (lane>>4)*8, so we need [p, p+8) and [p+16, p+24).
__device__ __forceinline__ v16bf load_frag16(const unsigned short* p) {
    v8bf lo = *reinterpret_cast<const v8bf*>(p);
    v8bf hi = *reinterpret_cast<const v8bf*>(p + 16);
    v16bf r;
#pragma unroll
    for (int i = 0; i < 8; ++i) { r[i] = lo[i]; r[i + 8] = hi[i]; }
    return r;
}

// ---------------------------------------------------------------------------
// Pack kernels
// ---------------------------------------------------------------------------
__global__ void cast_f32_to_bf16_kernel(const float* __restrict__ src,
                                        unsigned short* __restrict__ dst, int n) {
    int i = blockIdx.x * blockDim.x + threadIdx.x;
    if (i < n) dst[i] = f32_to_bf16(src[i]);
}

// W[z][K][Ncols] (f32, row-major) -> Wt[z][Npad][Kpad] (bf16, zero padded, transposed)
__global__ void pack_wt_kernel(const float* __restrict__ W, unsigned short* __restrict__ Wt,
                               int K, int Ncols, int Kpad, int Npad, int total) {
    int idx = blockIdx.x * blockDim.x + threadIdx.x;
    if (idx >= total) return;
    int k = idx % Kpad;
    int n = (idx / Kpad) % Npad;
    int z = idx / (Kpad * Npad);
    float v = 0.0f;
    if (k < K && n < Ncols) v = W[((long long)z * K + k) * Ncols + n];
    Wt[idx] = f32_to_bf16(v);
}

// b[z][Ncols] -> bp[z][Npad] (f32, zero padded)
__global__ void pack_bias_kernel(const float* __restrict__ b, float* __restrict__ bp,
                                 int Ncols, int Npad, int total) {
    int idx = blockIdx.x * blockDim.x + threadIdx.x;
    if (idx >= total) return;
    int n = idx % Npad;
    int z = idx / Npad;
    bp[idx] = (n < Ncols) ? b[(long long)z * Ncols + n] : 0.0f;
}

// ---------------------------------------------------------------------------
// Batched GEMM:  C[z] = act( A[z] (rows x K, bf16) * Bt[z]^T + bias[z] )
//   A  : row-major [rows, lda],  batch stride aBatch (elements)
//   Bt : row-major [Ncols, ldb] (i.e. B transposed), batch stride bBatch
//   C  : element (z,row,col) at  z*cBatch + row*ldc + col
// Block: 256 threads = 8 waves. Wave w computes a 64x32 tile:
// rows [bx*512 + w*64, +64), cols [by*32, +32) as a 4x2 grid of 16x16
// WMMA tiles -> 8 v_wmma per K-step from 6 fragment loads (12x b128).
// ---------------------------------------------------------------------------
__global__ void __launch_bounds__(256)
gemm_wmma_bf16_kernel(const unsigned short* __restrict__ A, long long aBatch, int lda,
                      const unsigned short* __restrict__ Bt, long long bBatch, int ldb,
                      const float* __restrict__ bias, long long biasBatch,
                      void* __restrict__ C, long long cBatch, int ldc,
                      int K, int relu, int storeBf16) {
    const int z    = blockIdx.z;
    const int wave = threadIdx.x >> 5;
    const int lane = threadIdx.x & 31;
    const int r0   = blockIdx.x * 512 + wave * 64;
    const int c0   = blockIdx.y * 32;

    const int rr   = lane & 15;
    const int ksel = (lane >> 4) * 8;   // K-chunk select per fragment layout

    const unsigned short* Az = A  + (long long)z * aBatch;
    const unsigned short* Bz = Bt + (long long)z * bBatch;

    const unsigned short* ap[4];
#pragma unroll
    for (int mi = 0; mi < 4; ++mi)
        ap[mi] = Az + (long long)(r0 + mi * 16 + rr) * lda + ksel;
    const unsigned short* bp0 = Bz + (long long)(c0 + rr)      * ldb + ksel;
    const unsigned short* bp1 = Bz + (long long)(c0 + 16 + rr) * ldb + ksel;

    v8f acc[4][2];
#pragma unroll
    for (int mi = 0; mi < 4; ++mi) { acc[mi][0] = {}; acc[mi][1] = {}; }

    for (int k0 = 0; k0 < K; k0 += 32) {
        // prefetch next K-block of the streamed operand (A); weights sit in L2
        if (k0 + 32 < K) {
            __builtin_prefetch(ap[0] + k0 + 32, 0, 3);
            __builtin_prefetch(ap[2] + k0 + 32, 0, 3);
        }
        v16bf b0 = load_frag16(bp0 + k0);
        v16bf b1 = load_frag16(bp1 + k0);
#pragma unroll
        for (int mi = 0; mi < 4; ++mi) {
            v16bf a = load_frag16(ap[mi] + k0);
            acc[mi][0] = __builtin_amdgcn_wmma_f32_16x16x32_bf16(
                false, a, false, b0, (short)0, acc[mi][0], false, false);
            acc[mi][1] = __builtin_amdgcn_wmma_f32_16x16x32_bf16(
                false, a, false, b1, (short)0, acc[mi][1], false, false);
        }
    }

    // C/D layout: lanes 0-15: M = vgpr, N = lane; lanes 16-31: M = vgpr + 8, N = lane - 16.
    const int g  = lane >> 4;
    const int nn = lane & 15;
    const float bz0 = bias[(long long)z * biasBatch + c0 + nn];
    const float bz1 = bias[(long long)z * biasBatch + c0 + 16 + nn];

    unsigned short* Cb = (unsigned short*)C;
    float*          Cf = (float*)C;

#pragma unroll
    for (int mi = 0; mi < 4; ++mi) {
#pragma unroll
        for (int r = 0; r < 8; ++r) {
            int row = r0 + mi * 16 + r + g * 8;
            float v0 = acc[mi][0][r] + bz0;
            float v1 = acc[mi][1][r] + bz1;
            if (relu) { v0 = fmaxf(v0, 0.0f); v1 = fmaxf(v1, 0.0f); }
            long long i0 = (long long)z * cBatch + (long long)row * ldc + (c0 + nn);
            long long i1 = (long long)z * cBatch + (long long)row * ldc + (c0 + 16 + nn);
            if (storeBf16) {
                Cb[i0] = f32_to_bf16(v0); Cb[i1] = f32_to_bf16(v1);
            } else {
                Cf[i0] = v0; Cf[i1] = v1;
            }
        }
    }
}

// ---------------------------------------------------------------------------
// Gating: gated[n][b][f] = sum_m gw[n,m] * (u[b,n,m] < sigmoid(gl[n,m])) * eo[m][b][f]
// One block per sample b; 256 threads cover F=512 (2 per thread).
// ---------------------------------------------------------------------------
__global__ void __launch_bounds__(256)
gating_kernel(const float* __restrict__ u, const float* __restrict__ gw,
              const float* __restrict__ gl, const unsigned short* __restrict__ eo,
              unsigned short* __restrict__ gated) {
    __shared__ float w[MOE_N * MOE_M];   // 128 effective gate weights for this sample
    const int b = blockIdx.x;
    const int t = threadIdx.x;
    if (t < MOE_N * MOE_M) {
        float prob = 1.0f / (1.0f + expf(-gl[t]));
        float bit  = (u[(long long)b * (MOE_N * MOE_M) + t] < prob) ? 1.0f : 0.0f;
        w[t] = gw[t] * bit;
    }
    __syncthreads();

    for (int f = t; f < 512; f += 256) {
        float e[MOE_M];
#pragma unroll
        for (int m = 0; m < MOE_M; ++m)
            e[m] = bf16_to_f32(eo[(long long)m * MOE_B * 512 + (long long)b * 512 + f]);
#pragma unroll
        for (int n = 0; n < MOE_N; ++n) {
            float acc = 0.0f;
#pragma unroll
            for (int m = 0; m < MOE_M; ++m) acc += w[n * MOE_M + m] * e[m];
            gated[(long long)n * MOE_B * 512 + (long long)b * 512 + f] = f32_to_bf16(acc);
        }
    }
}

// logits_loss[n] = B * sum_m log_sigmoid(gl[n,m])
__global__ void loss_kernel(const float* __restrict__ gl, float* __restrict__ out) {
    int n = threadIdx.x;
    if (n < MOE_N) {
        float s = 0.0f;
        for (int m = 0; m < MOE_M; ++m) {
            float x = gl[n * MOE_M + m];
            s += fminf(x, 0.0f) - log1pf(expf(-fabsf(x)));  // stable log_sigmoid
        }
        out[n] = (float)MOE_B * s;
    }
}

// ---------------------------------------------------------------------------
// Host-side orchestration
// ---------------------------------------------------------------------------
extern "C" void kernel_launch(void* const* d_in, const int* in_sizes, int n_in,
                              void* d_out, int out_size, void* d_ws, size_t ws_size,
                              hipStream_t stream) {
    (void)in_sizes; (void)n_in; (void)out_size; (void)ws_size;

    const float* x   = (const float*)d_in[0];
    // d_in[1] = task (unused by the reference output)
    const float* u   = (const float*)d_in[2];
    const float* ew1 = (const float*)d_in[3];
    const float* eb1 = (const float*)d_in[4];
    const float* ew2 = (const float*)d_in[5];
    const float* eb2 = (const float*)d_in[6];
    const float* ew3 = (const float*)d_in[7];
    const float* eb3 = (const float*)d_in[8];
    const float* gw  = (const float*)d_in[9];
    const float* gl  = (const float*)d_in[10];
    const float* tw1 = (const float*)d_in[11];
    const float* tb1 = (const float*)d_in[12];
    const float* tw2 = (const float*)d_in[13];
    const float* tb2 = (const float*)d_in[14];
    const float* tw3 = (const float*)d_in[15];
    const float* tb3 = (const float*)d_in[16];

    const int B = MOE_B, M = MOE_M, N = MOE_N;

    // ---- workspace carve-out (256B aligned) ----
    size_t off = 0;
    auto carve = [&](size_t bytes) -> void* {
        void* p = (char*)d_ws + off;
        off += (bytes + 255) & ~(size_t)255;
        return p;
    };
    unsigned short* xb    = (unsigned short*)carve((size_t)B * 1024 * 2);
    unsigned short* ew1t  = (unsigned short*)carve((size_t)M * 128 * 1024 * 2);
    float*          eb1p  = (float*)         carve((size_t)M * 128 * 4);
    unsigned short* ew2t  = (unsigned short*)carve((size_t)M * 96 * 128 * 2);
    float*          eb2p  = (float*)         carve((size_t)M * 96 * 4);
    unsigned short* ew3t  = (unsigned short*)carve((size_t)M * 512 * 96 * 2);
    float*          eb3p  = (float*)         carve((size_t)M * 512 * 4);
    unsigned short* tw1t  = (unsigned short*)carve((size_t)N * 128 * 512 * 2);
    float*          tb1p  = (float*)         carve((size_t)N * 128 * 4);
    unsigned short* tw2t  = (unsigned short*)carve((size_t)N * 96 * 128 * 2);
    float*          tb2p  = (float*)         carve((size_t)N * 96 * 4);
    unsigned short* tw3t  = (unsigned short*)carve((size_t)N * 256 * 96 * 2);
    float*          tb3p  = (float*)         carve((size_t)N * 256 * 4);
    unsigned short* h1    = (unsigned short*)carve((size_t)M * B * 128 * 2);
    unsigned short* h2    = (unsigned short*)carve((size_t)M * B * 96 * 2);
    unsigned short* eo    = (unsigned short*)carve((size_t)M * B * 512 * 2);
    unsigned short* gated = (unsigned short*)carve((size_t)N * B * 512 * 2);
    unsigned short* t1    = (unsigned short*)carve((size_t)N * B * 128 * 2);
    unsigned short* t2    = (unsigned short*)carve((size_t)N * B * 96 * 2);

    // ---- pack inputs to bf16 (transposed + padded weights) ----
    {
        int n = B * 1024;
        cast_f32_to_bf16_kernel<<<(n + 255) / 256, 256, 0, stream>>>(x, xb, n);
    }
    auto packW = [&](const float* W, unsigned short* Wt, int Z, int K, int Nc, int Kp, int Np) {
        int total = Z * Np * Kp;
        pack_wt_kernel<<<(total + 255) / 256, 256, 0, stream>>>(W, Wt, K, Nc, Kp, Np, total);
    };
    auto packB = [&](const float* bsrc, float* bpd, int Z, int Nc, int Np) {
        int total = Z * Np;
        pack_bias_kernel<<<(total + 255) / 256, 256, 0, stream>>>(bsrc, bpd, Nc, Np, total);
    };
    packW(ew1, ew1t, M, 1024, 120, 1024, 128);  packB(eb1, eb1p, M, 120, 128);
    packW(ew2, ew2t, M, 120,  84,  128,  96);   packB(eb2, eb2p, M, 84,  96);
    packW(ew3, ew3t, M, 84,   512, 96,   512);  packB(eb3, eb3p, M, 512, 512);
    packW(tw1, tw1t, N, 512,  120, 512,  128);  packB(tb1, tb1p, N, 120, 128);
    packW(tw2, tw2t, N, 120,  84,  128,  96);   packB(tb2, tb2p, N, 84,  96);
    packW(tw3, tw3t, N, 84,   256, 96,   256);  packB(tb3, tb3p, N, 256, 256);

    const dim3 blk(256);
    const int gx = B / 512;  // 16 row-tiles of 512 (8 waves x 64 rows)

    // ---- expert MLP (batched over M experts via grid.z) ----
    gemm_wmma_bf16_kernel<<<dim3(gx, 128 / 32, M), blk, 0, stream>>>(
        xb, 0LL, 1024, ew1t, 128LL * 1024, 1024, eb1p, 128,
        h1, (long long)B * 128, 128, 1024, /*relu=*/1, /*bf16=*/1);
    gemm_wmma_bf16_kernel<<<dim3(gx, 96 / 32, M), blk, 0, stream>>>(
        h1, (long long)B * 128, 128, ew2t, 96LL * 128, 128, eb2p, 96,
        h2, (long long)B * 96, 96, 128, 1, 1);
    gemm_wmma_bf16_kernel<<<dim3(gx, 512 / 32, M), blk, 0, stream>>>(
        h2, (long long)B * 96, 96, ew3t, 512LL * 96, 96, eb3p, 512,
        eo, (long long)B * 512, 512, 96, 0, 1);

    // ---- stochastic gating mix ----
    gating_kernel<<<dim3(B), blk, 0, stream>>>(u, gw, gl, eo, gated);

    // ---- task heads (batched over N tasks via grid.z) ----
    gemm_wmma_bf16_kernel<<<dim3(gx, 128 / 32, N), blk, 0, stream>>>(
        gated, (long long)B * 512, 512, tw1t, 128LL * 512, 512, tb1p, 128,
        t1, (long long)B * 128, 128, 512, 1, 1);
    gemm_wmma_bf16_kernel<<<dim3(gx, 96 / 32, N), blk, 0, stream>>>(
        t1, (long long)B * 128, 128, tw2t, 96LL * 128, 128, tb2p, 96,
        t2, (long long)B * 96, 96, 128, 1, 1);
    // final layer writes f32 directly into d_out[B, N, 256]:
    //   element (z=n, row=b, col) at n*256 + b*(N*256) + col
    gemm_wmma_bf16_kernel<<<dim3(gx, 256 / 32, N), blk, 0, stream>>>(
        t2, (long long)B * 96, 96, tw3t, 256LL * 96, 96, tb3p, 256,
        d_out, 256LL, N * 256, 96, 0, /*bf16=*/0);

    // ---- logits loss appended after the output tensor ----
    loss_kernel<<<1, 32, 0, stream>>>(gl, (float*)d_out + (size_t)B * N * 256);
}
